// ChunkwiseSSMLayer_62302795596578
// MI455X (gfx1250) — matrix operand order repaired and checked by reading, not compiled, not run on hardware
//
#include <hip/hip_runtime.h>
#include <hip/hip_bf16.h>

typedef __attribute__((ext_vector_type(16))) __bf16 v16bf;
typedef __attribute__((ext_vector_type(8)))  __bf16 v8bf;
typedef __attribute__((ext_vector_type(4)))  __bf16 v4bf;
typedef __attribute__((ext_vector_type(8)))  float  v8f;
typedef __attribute__((ext_vector_type(4)))  int    v4i;

#define DEV __device__ __forceinline__

static constexpr int Bb  = 4;
static constexpr int S   = 4096;
static constexpr int D   = 1024;
static constexpr int N   = 64;
static constexpr int CH  = 16;
static constexpr int KCH = S / CH;       // 256 chunks per batch row
static constexpr int M   = Bb * S;       // 16384 rows
static constexpr int KSTEP = 128;        // K staged per LDS step
static constexpr int LDSLD = KSTEP + 8;  // padded LDS row: 272B stride, conflict-free
static constexpr int NSTG  = D / KSTEP;  // 8 stages

// ---------- CDNA5 async global->LDS path (guarded; falls back to sync staging) ----
#if __has_builtin(__builtin_amdgcn_global_load_async_to_lds_b128) && \
    __has_builtin(__builtin_amdgcn_s_wait_asynccnt)
#define HAS_ASYNC_LDS 1
typedef __attribute__((address_space(1))) v4i gv4i;   // global int4*
typedef __attribute__((address_space(3))) v4i lv4i;   // LDS int4*
DEV void async_copy_b128(const __bf16* g, __bf16* l) {
  __builtin_amdgcn_global_load_async_to_lds_b128((gv4i*)(g), (lv4i*)(l), 0, 0);
}
#else
#define HAS_ASYNC_LDS 0
#endif

// ---------- WMMA fragment loaders (CDNA5 wave32 layouts) ----------
// A-matrix 16x32 bf16: lane m = lane&15; holds K=[koff..koff+7] and
// K=[koff+16..koff+23] where koff = (lane>>4)*8.
DEV v16bf load_a_frag(const __bf16* p) {
  v8bf lo = *(const v8bf*)(p);
  v8bf hi = *(const v8bf*)(p + 16);
  v16bf f;
#pragma unroll
  for (int i = 0; i < 8; ++i) { f[i] = lo[i]; f[i + 8] = hi[i]; }
  return f;
}
// B-matrix 32x16 bf16 from row-major B^T (N x K): lane n = lane&15 holds
// K=[koff..koff+7] and K=[koff+16..koff+23] where koff = (lane>>4)*8.
DEV v16bf load_b_frag(const __bf16* p) {
  v8bf lo = *(const v8bf*)(p);
  v8bf hi = *(const v8bf*)(p + 16);
  v16bf f;
#pragma unroll
  for (int i = 0; i < 8; ++i) { f[i] = lo[i]; f[i + 8] = hi[i]; }
  return f;
}

DEV float sigmoidf_(float v) { return 1.0f / (1.0f + __expf(-v)); }

DEV v8f wmma_bf16(v16bf a, v16bf b, v8f c) {
  return __builtin_amdgcn_wmma_f32_16x16x32_bf16(false, a, false, b, (short)0, c,
                                                 false, false);
}

// ---------- elementwise conversion kernels ----------
__global__ void k_cvt_x(const float* __restrict__ x, __bf16* __restrict__ xb) {
  int i = (blockIdx.x * blockDim.x + threadIdx.x) * 4;
  float4 v = *(const float4*)(x + i);
  v4bf o;
  o[0] = (__bf16)v.x; o[1] = (__bf16)v.y; o[2] = (__bf16)v.z; o[3] = (__bf16)v.w;
  *(v4bf*)(xb + i) = o;
}

// out[c*rows + r] = (bf16) in[r*cols + c]   (store W^T for WMMA B operand)
__global__ void k_transpose_bf16(const float* __restrict__ in, __bf16* __restrict__ out,
                                 int rows, int cols) {
  int idx = blockIdx.x * blockDim.x + threadIdx.x;
  if (idx >= rows * cols) return;
  int r = idx / cols, c = idx - r * cols;
  out[(size_t)c * rows + r] = (__bf16)in[idx];
}

// =====================================================================
// Shared block-GEMM machinery: block tile 128 rows x 64 cols, 8 warps
// (each warp 16 rows x 64 cols, 4 accumulators), K staged 128/step into
// double-buffered, row-padded LDS via async global->LDS DMA.
// =====================================================================

// ---------- projection GEMM: [gates | b_all] = x @ [gate_W | B_W] ----------
__global__ void __launch_bounds__(256)
k_proj(const __bf16* __restrict__ xb, const __bf16* __restrict__ wgbT,
       const float* __restrict__ gate_b,
       float* __restrict__ a_buf, float* __restrict__ b_buf) {
  __shared__ __bf16 sA[2][128 * LDSLD];   // 2 x 34 KB
  __shared__ __bf16 sB[2][64 * LDSLD];    // 2 x 17 KB
  const int tid  = threadIdx.x;
  const int lane = tid & 31;
  const int warp = tid >> 5;
  const int m0 = (blockIdx.x >> 1) * 128;   // row block
  const int j0 = (blockIdx.x & 1) * 64;     // col block: 0 = gates, 64 = b_all
  const int ABUF = 128 * LDSLD, BBUF = 64 * LDSLD;

  // staging: A 32 KB -> 128 B/thread (8xb128); B 16 KB -> 64 B/thread (4xb128)
  const int arow = tid >> 1, acol = (tid & 1) * 64;
  const int brow = tid >> 2, bcol = (tid & 3) * 32;
  const __bf16* gA = xb   + (size_t)(m0 + arow) * D + acol;
  const __bf16* gB = wgbT + (size_t)(j0 + brow) * D + bcol;
  __bf16* lA = &sA[0][0] + arow * LDSLD + acol;
  __bf16* lB = &sB[0][0] + brow * LDSLD + bcol;

  v8f acc[4] = {};
  const int am = lane & 15, ak = (lane >> 4) << 3;
  const int bn = lane & 15, bk = (lane >> 4) << 3;
  const __bf16* aBase = &sA[0][0] + (warp * 16 + am) * LDSLD + ak;
  const __bf16* bBase = &sB[0][0] + bn * LDSLD + bk;

  auto compute = [&](int cb) {
    const __bf16* aB = aBase + cb * ABUF;
    const __bf16* bB = bBase + cb * BBUF;
#pragma unroll
    for (int ks = 0; ks < KSTEP; ks += 32) {
      v16bf af = load_a_frag(aB + ks);
      v16bf bfr[4];
#pragma unroll
      for (int j = 0; j < 4; ++j) bfr[j] = load_b_frag(bB + j * 16 * LDSLD + ks);
#pragma unroll
      for (int j = 0; j < 4; ++j) acc[j] = wmma_bf16(af, bfr[j], acc[j]);
    }
  };

#if HAS_ASYNC_LDS
  auto issue = [&](int s) {
    const int kk = s * KSTEP;
    __bf16* la = lA + (s & 1) * ABUF;
    __bf16* lb = lB + (s & 1) * BBUF;
#pragma unroll
    for (int j = 0; j < 8; ++j) async_copy_b128(gA + kk + j * 8, la + j * 8);
#pragma unroll
    for (int j = 0; j < 4; ++j) async_copy_b128(gB + kk + j * 8, lb + j * 8);
  };
  issue(0);
  for (int s = 0; s < NSTG - 1; ++s) {
    issue(s + 1);
    __builtin_amdgcn_s_wait_asynccnt(12);  // stage s complete; s+1 in flight
    __syncthreads();
    compute(s & 1);
    __syncthreads();                       // all warps done before s+2 overwrites
  }
  __builtin_amdgcn_s_wait_asynccnt(0);
  __syncthreads();
  compute((NSTG - 1) & 1);
#else
  for (int s = 0; s < NSTG; ++s) {
    const int kk = s * KSTEP;
#pragma unroll
    for (int j = 0; j < 8; ++j) *(uint4*)(lA + j * 8) = *(const uint4*)(gA + kk + j * 8);
#pragma unroll
    for (int j = 0; j < 4; ++j) *(uint4*)(lB + j * 8) = *(const uint4*)(gB + kk + j * 8);
    __syncthreads();
    compute(0);
    __syncthreads();
  }
#endif

  // C/D layout: row = r + 8*(lane>=16), col = tile*16 + (lane&15)
  const int coln = lane & 15, rowadd = (lane >> 4) << 3;
  const int wm0 = m0 + warp * 16;
  if (j0 == 0) {   // gate columns: sigmoid(v + bias)
#pragma unroll
    for (int j = 0; j < 4; ++j) {
      int jc = (j << 4) + coln;
      float gb = gate_b[jc];
#pragma unroll
      for (int r = 0; r < 8; ++r)
        a_buf[(size_t)(wm0 + r + rowadd) * N + jc] = sigmoidf_(acc[j][r] + gb);
    }
  } else {         // b_all columns: raw
#pragma unroll
    for (int j = 0; j < 4; ++j) {
      int jc = (j << 4) + coln;
#pragma unroll
      for (int r = 0; r < 8; ++r)
        b_buf[(size_t)(wm0 + r + rowadd) * N + jc] = acc[j][r];
    }
  }
}

// ---------- phase A: per-chunk local scan + prefix products + carries ----------
__global__ void k_chunk_local(const float* __restrict__ a_buf, const float* __restrict__ b_buf,
                              float* __restrict__ hloc, float* __restrict__ pbuf,
                              float* __restrict__ Ac, float* __restrict__ fc) {
  int tid = blockIdx.x * blockDim.x + threadIdx.x;   // Bb*KCH*N = 65536
  int n = tid & (N - 1);
  int c = tid >> 6;                                  // global chunk id = b*KCH + k
  float h = 0.f, p = 1.f;
  size_t base = (size_t)c * CH * N + n;
#pragma unroll
  for (int t = 0; t < CH; ++t) {
    size_t idx = base + (size_t)t * N;
    float av = a_buf[idx], bv = b_buf[idx];
    h = fmaf(av, h, bv);
    p *= av;
    hloc[idx] = h;
    pbuf[idx] = p;
  }
  Ac[tid] = p;
  fc[tid] = h;
}

// ---------- phase B: inter-chunk carry recurrence (256 serial steps) ----------
__global__ void k_carry_scan(const float* __restrict__ Ac, const float* __restrict__ fc,
                             float* __restrict__ hin) {
  int tid = threadIdx.x;            // Bb*N = 256
  int n = tid & (N - 1);
  int b = tid >> 6;
  float h = 0.f;
  for (int k = 0; k < KCH; ++k) {
    int idx = (b * KCH + k) * N + n;
    hin[idx] = h;
    h = fmaf(Ac[idx], h, fc[idx]);
  }
}

// ---------- phase C: h = hloc + p * h_in, emit bf16 ----------
__global__ void k_finalize_h(const float* __restrict__ hloc, const float* __restrict__ pbuf,
                             const float* __restrict__ hin, __bf16* __restrict__ hb) {
  int idx = blockIdx.x * blockDim.x + threadIdx.x;   // M*N = 1M
  int n = idx & (N - 1);
  int m = idx >> 6;
  int c = m >> 4;
  float h = hloc[idx] + pbuf[idx] * hin[c * N + n];
  hb[idx] = (__bf16)h;
}

// ---------- fused output: out = (h @ C_W) * sigmoid(x @ out_W) ----------
__global__ void __launch_bounds__(256)
k_final(const __bf16* __restrict__ xb, const __bf16* __restrict__ owT,
        const __bf16* __restrict__ hb, const __bf16* __restrict__ cwT,
        float* __restrict__ out) {
  __shared__ __bf16 sA[2][128 * LDSLD];   // 2 x 34 KB
  __shared__ __bf16 sB[2][64 * LDSLD];    // 2 x 17 KB
  const int tid  = threadIdx.x;
  const int lane = tid & 31;
  const int warp = tid >> 5;
  const int m0 = (blockIdx.x >> 4) * 128;  // 128 row-blocks
  const int e0 = (blockIdx.x & 15) * 64;   // 16 col-blocks
  const int ABUF = 128 * LDSLD, BBUF = 64 * LDSLD;

  const int arow = tid >> 1, acol = (tid & 1) * 64;
  const int brow = tid >> 2, bcol = (tid & 3) * 32;
  const __bf16* gA = xb  + (size_t)(m0 + arow) * D + acol;
  const __bf16* gB = owT + (size_t)(e0 + brow) * D + bcol;
  __bf16* lA = &sA[0][0] + arow * LDSLD + acol;
  __bf16* lB = &sB[0][0] + brow * LDSLD + bcol;

  v8f accg[4] = {};
  const int am = lane & 15, ak = (lane >> 4) << 3;
  const int bn = lane & 15, bk = (lane >> 4) << 3;
  const __bf16* aBase = &sA[0][0] + (warp * 16 + am) * LDSLD + ak;
  const __bf16* bBase = &sB[0][0] + bn * LDSLD + bk;

  auto compute = [&](int cb) {
    const __bf16* aB = aBase + cb * ABUF;
    const __bf16* bB = bBase + cb * BBUF;
#pragma unroll
    for (int ks = 0; ks < KSTEP; ks += 32) {
      v16bf af = load_a_frag(aB + ks);
      v16bf bfr[4];
#pragma unroll
      for (int j = 0; j < 4; ++j) bfr[j] = load_b_frag(bB + j * 16 * LDSLD + ks);
#pragma unroll
      for (int j = 0; j < 4; ++j) accg[j] = wmma_bf16(af, bfr[j], accg[j]);
    }
  };

#if HAS_ASYNC_LDS
  auto issue = [&](int s) {
    const int kk = s * KSTEP;
    __bf16* la = lA + (s & 1) * ABUF;
    __bf16* lb = lB + (s & 1) * BBUF;
#pragma unroll
    for (int j = 0; j < 8; ++j) async_copy_b128(gA + kk + j * 8, la + j * 8);
#pragma unroll
    for (int j = 0; j < 4; ++j) async_copy_b128(gB + kk + j * 8, lb + j * 8);
  };
  issue(0);
  for (int s = 0; s < NSTG - 1; ++s) {
    issue(s + 1);
    __builtin_amdgcn_s_wait_asynccnt(12);
    __syncthreads();
    compute(s & 1);
    __syncthreads();
  }
  __builtin_amdgcn_s_wait_asynccnt(0);
  __syncthreads();
  compute((NSTG - 1) & 1);
#else
  for (int s = 0; s < NSTG; ++s) {
    const int kk = s * KSTEP;
#pragma unroll
    for (int j = 0; j < 8; ++j) *(uint4*)(lA + j * 8) = *(const uint4*)(gA + kk + j * 8);
#pragma unroll
    for (int j = 0; j < 4; ++j) *(uint4*)(lB + j * 8) = *(const uint4*)(gB + kk + j * 8);
    __syncthreads();
    compute(0);
    __syncthreads();
  }
#endif

  // y GEMM: K = 64, tiny -> direct from global
  v8f accy[4] = {};
  const int wm0 = m0 + warp * 16;
  const __bf16* hrow  = hb  + (size_t)(wm0 + am) * N + ak;
  const __bf16* yrowb = cwT + (size_t)(e0 + bn) * N + bk;
#pragma unroll
  for (int k = 0; k < N; k += 32) {
    v16bf af = load_a_frag(hrow + k);
    v16bf bfr[4];
#pragma unroll
    for (int j = 0; j < 4; ++j) bfr[j] = load_b_frag(yrowb + (size_t)(j * 16) * N + k);
#pragma unroll
    for (int j = 0; j < 4; ++j) accy[j] = wmma_bf16(af, bfr[j], accy[j]);
  }

  const int coln = lane & 15, rowadd = (lane >> 4) << 3;
#pragma unroll
  for (int j = 0; j < 4; ++j) {
#pragma unroll
    for (int r = 0; r < 8; ++r) {
      int row = wm0 + r + rowadd;
      int col = e0 + (j << 4) + coln;
      out[(size_t)row * D + col] = accy[j][r] * sigmoidf_(accg[j][r]);
    }
  }
}

extern "C" void kernel_launch(void* const* d_in, const int* in_sizes, int n_in,
                              void* d_out, int out_size, void* d_ws, size_t ws_size,
                              hipStream_t stream) {
  const float* x      = (const float*)d_in[0];   // (4,4096,1024)
  const float* gate_W = (const float*)d_in[1];   // (1024,64)
  const float* gate_b = (const float*)d_in[2];   // (64,)
  const float* B_W    = (const float*)d_in[3];   // (1024,64)
  const float* C_W    = (const float*)d_in[4];   // (64,1024)
  const float* out_W  = (const float*)d_in[5];   // (1024,1024)
  // d_in[6] = mix_weight: inter-chunk mixing algebraically cancels (h_next == h_final)
  // d_in[7] = chunk_size (16, hardcoded)
  float* out = (float*)d_out;

  char* ws = (char*)d_ws;
  size_t off = 0;
  auto alloc = [&](size_t bytes) {
    void* p = ws + off;
    off = (off + bytes + 255) & ~(size_t)255;
    return p;
  };
  __bf16* xb   = (__bf16*)alloc((size_t)M * D * 2);       // 32 MB
  __bf16* wgbT = (__bf16*)alloc((size_t)2 * N * D * 2);   // [gate_W^T ; B_W^T]
  __bf16* owT  = (__bf16*)alloc((size_t)D * D * 2);       // out_W^T
  __bf16* cwT  = (__bf16*)alloc((size_t)D * N * 2);       // C_W^T
  float* a_buf = (float*)alloc((size_t)M * N * 4);
  float* b_buf = (float*)alloc((size_t)M * N * 4);
  float* hloc  = (float*)alloc((size_t)M * N * 4);
  float* pbuf  = (float*)alloc((size_t)M * N * 4);
  __bf16* hb   = (__bf16*)alloc((size_t)M * N * 2);
  float* Ac    = (float*)alloc((size_t)Bb * KCH * N * 4);
  float* fc    = (float*)alloc((size_t)Bb * KCH * N * 4);
  float* hin   = (float*)alloc((size_t)Bb * KCH * N * 4);
  (void)ws_size; (void)in_sizes; (void)n_in; (void)out_size;

  k_cvt_x<<<(M * D / 4) / 256, 256, 0, stream>>>(x, xb);
  k_transpose_bf16<<<(D * N + 255) / 256, 256, 0, stream>>>(gate_W, wgbT, D, N);
  k_transpose_bf16<<<(D * N + 255) / 256, 256, 0, stream>>>(B_W, wgbT + (size_t)N * D, D, N);
  k_transpose_bf16<<<(D * D + 255) / 256, 256, 0, stream>>>(out_W, owT, D, D);
  k_transpose_bf16<<<(N * D + 255) / 256, 256, 0, stream>>>(C_W, cwT, N, D);
  k_proj<<<(M / 128) * 2, 256, 0, stream>>>(xb, wgbT, gate_b, a_buf, b_buf);
  k_chunk_local<<<(Bb * KCH * N) / 256, 256, 0, stream>>>(a_buf, b_buf, hloc, pbuf, Ac, fc);
  k_carry_scan<<<1, Bb * N, 0, stream>>>(Ac, fc, hin);
  k_finalize_h<<<(M * N) / 256, 256, 0, stream>>>(hloc, pbuf, hin, hb);
  k_final<<<(M / 128) * (D / 64), 256, 0, stream>>>(xb, owT, hb, cwT, out);
}
